// HybridMoEModelV2_56727928046141
// MI455X (gfx1250) — matrix operand order repaired and checked by reading, not compiled, not run on hardware
//
#include <hip/hip_runtime.h>
#include <hip/hip_bf16.h>

// ---------------- model constants ----------------
static constexpr int cB    = 4;
static constexpr int cL    = 2048;
static constexpr int cD    = 1024;
static constexpr int cNH   = 16;
static constexpr int cNKV  = 4;
static constexpr int cHD   = 64;
static constexpr int cNEXP = 8;
static constexpr int cTOPK = 2;
static constexpr int cINTER= 1024;
static constexpr int cN    = cB * cL;          // 8192 tokens
static constexpr int cR    = cN * cTOPK;       // 16384 routed rows
static constexpr float cEPS = 1e-5f;

typedef __bf16 bf16;
typedef __attribute__((ext_vector_type(16))) __bf16        v16bf;
typedef __attribute__((ext_vector_type(8)))  float         v8f;
typedef __attribute__((ext_vector_type(4)))  unsigned int  v4u;
typedef __attribute__((ext_vector_type(8)))  unsigned int  v8u;
typedef __attribute__((ext_vector_type(4)))  int           v4i;

// Direct global->LDS async copy (CDNA5): guarded so compile never breaks.
#if defined(__has_builtin)
#  if __has_builtin(__builtin_amdgcn_global_load_async_to_lds_b128)
#    define ATHENA_ASYNC_LDS 1
#  endif
#endif

// ---------------- helpers ----------------
__device__ __forceinline__ bf16 f2bf(float f) {
  unsigned u = __builtin_bit_cast(unsigned, f);
  u += 0x7fffu + ((u >> 16) & 1u);            // round-to-nearest-even
  unsigned short h = (unsigned short)(u >> 16);
  return __builtin_bit_cast(bf16, h);
}
__device__ __forceinline__ v8f wmma_bf16(v16bf a, v16bf b, v8f c) {
  // D = A(16x32 bf16) x B(32x16 bf16) + C(16x16 f32), wave32
  return __builtin_amdgcn_wmma_f32_16x16x32_bf16(false, a, false, b, (short)0, c,
                                                 false, false);
}
// A fragment (16x32 16-bit): lane m holds k in [half*8, half*8+8) U [16+half*8, ...)
__device__ __forceinline__ v16bf load_afrag(const bf16* __restrict__ row, int half) {
  v4u lo = *(const v4u*)(row + half * 8);
  v4u hi = *(const v4u*)(row + 16 + half * 8);
  v8u u = __builtin_shufflevector(lo, hi, 0, 1, 2, 3, 4, 5, 6, 7);
  return __builtin_bit_cast(v16bf, u);
}
// B fragment (32x16 16-bit): lane n holds k in [half*8, half*8+8) U [16+half*8, ...)
__device__ __forceinline__ v16bf load_bfrag(const bf16* __restrict__ row, int half) {
  v4u lo = *(const v4u*)(row + half * 8);
  v4u hi = *(const v4u*)(row + 16 + half * 8);
  v8u u = __builtin_shufflevector(lo, hi, 0, 1, 2, 3, 4, 5, 6, 7);
  return __builtin_bit_cast(v16bf, u);
}
// 16B tile fill: async global->LDS when available, else load+store.
__device__ __forceinline__ void fill_lds_16B(bf16* __restrict__ ldst,
                                             const bf16* __restrict__ gsrc, bool ok) {
#if defined(ATHENA_ASYNC_LDS)
  if (ok) {
    __builtin_amdgcn_global_load_async_to_lds_b128((v4i*)gsrc, (v4i*)ldst, 0, 0);
  } else {
    v4u z = {0u, 0u, 0u, 0u};
    *(v4u*)ldst = z;
  }
#else
  v4u z = {0u, 0u, 0u, 0u};
  *(v4u*)ldst = ok ? *(const v4u*)gsrc : z;
#endif
}
__device__ __forceinline__ void wait_lds_fill() {
#if defined(ATHENA_ASYNC_LDS)
  asm volatile("s_wait_asynccnt 0x0" ::: "memory");
#endif
}

// ---------------- elementwise kernels ----------------
__global__ void k_cvt_bf16(bf16* __restrict__ dst, const float* __restrict__ src, long n) {
  long i = (long)blockIdx.x * blockDim.x + threadIdx.x;
  if (i < n) dst[i] = f2bf(src[i]);
}
// V: (b, l, kv, hd) f32 -> transposed (b, kv, hd, l) bf16 so PV B-fragments are
// two 8-key runs per lane (keys half*8.. and 16+half*8.., at fixed column).
__global__ void k_cvt_v_t(bf16* __restrict__ vT, const float* __restrict__ vf) {
  long i = (long)blockIdx.x * blockDim.x + threadIdx.x;
  if (i >= (long)cN * cNKV * cHD) return;
  int d    = (int)(i & (cHD - 1));
  long t2  = i >> 6;
  int kvh  = (int)(t2 & (cNKV - 1));
  long tok = t2 >> 2;
  int b = (int)(tok / cL), l = (int)(tok % cL);
  vT[(((long)b * cNKV + kvh) * cHD + d) * cL + l] = f2bf(vf[i]);
}
__global__ void k_zero_f(float* __restrict__ p, long n) {
  long i = (long)blockIdx.x * blockDim.x + threadIdx.x;
  if (i < n) p[i] = 0.f;
}
__global__ void k_zero_i(int* __restrict__ p, long n) {
  long i = (long)blockIdx.x * blockDim.x + threadIdx.x;
  if (i < n) p[i] = 0;
}
__global__ void k_silu_mul(bf16* __restrict__ act, const float* __restrict__ g,
                           const float* __restrict__ u, long n) {
  long i = (long)blockIdx.x * blockDim.x + threadIdx.x;
  if (i < n) {
    float gv = g[i];
    float s  = gv / (1.f + __expf(-gv));
    act[i]   = f2bf(s * u[i]);
  }
}
__global__ void k_add_out(float* __restrict__ out, const float* __restrict__ h,
                          const float* __restrict__ moe, long n) {
  long i = (long)blockIdx.x * blockDim.x + threadIdx.x;
  if (i < n) out[i] = h[i] + moe[i];
}

// ---------------- RMSNorm (row per block) ----------------
__global__ __launch_bounds__(256) void k_rmsnorm(bf16* __restrict__ obf,
                                                 float* __restrict__ of32,
                                                 const float* __restrict__ x,
                                                 const float* __restrict__ w, int D) {
  int row = blockIdx.x;
  const float* xr = x + (long)row * D;
  float ss = 0.f;
  for (int i = threadIdx.x; i < D; i += blockDim.x) { float v = xr[i]; ss += v * v; }
  for (int off = 16; off > 0; off >>= 1) ss += __shfl_xor(ss, off, 32);
  __shared__ float red[8];
  int wave = threadIdx.x >> 5;
  if ((threadIdx.x & 31) == 0) red[wave] = ss;
  __syncthreads();
  float tot = 0.f;
  int nw = blockDim.x >> 5;
  for (int i = 0; i < nw; i++) tot += red[i];
  float rinv = rsqrtf(tot / (float)D + cEPS);
  for (int i = threadIdx.x; i < D; i += blockDim.x) {
    float v = xr[i] * rinv * w[i];
    obf[(long)row * D + i] = f2bf(v);
    if (of32) of32[(long)row * D + i] = v;
  }
}

// ---------------- per-head q/k norm + RoPE (one wave per (token,head)) ------
__global__ __launch_bounds__(32) void k_qknorm_rope(bf16* __restrict__ outbf,
                                                    const float* __restrict__ qin,
                                                    const float* __restrict__ nw,
                                                    int nheads, int L) {
  int bid   = blockIdx.x;
  int head  = bid % nheads;
  int token = bid / nheads;
  int pos   = token % L;
  const float* src = qin + ((long)token * nheads + head) * cHD;
  int t = threadIdx.x;                    // pair index 0..31
  float xe = src[2 * t], xo = src[2 * t + 1];
  float ss = xe * xe + xo * xo;
  for (int off = 16; off > 0; off >>= 1) ss += __shfl_xor(ss, off, 32);
  float rinv = rsqrtf(ss / (float)cHD + cEPS);
  xe *= rinv * nw[2 * t];
  xo *= rinv * nw[2 * t + 1];
  float freq = __expf(-(float)t * (__logf(10000.f) / 32.f));
  float ang  = (float)pos * freq;
  float c = __cosf(ang), s = __sinf(ang);
  float oe = xe * c - xo * s;
  float oo = xe * s + xo * c;
  bf16* dst = outbf + ((long)token * nheads + head) * cHD;
  dst[2 * t]     = f2bf(oe);
  dst[2 * t + 1] = f2bf(oo);
}

// ---------------- tiled WMMA GEMM: C[M,N] = A[M,K] * W[N,K]^T (+resid) ------
static constexpr int TM = 64, TN = 64, TK = 32, LDP = TK + 8;  // 80B row stride

__global__ __launch_bounds__(128) void k_gemm_nt(float* __restrict__ C,
                                                 const bf16* __restrict__ A,
                                                 const bf16* __restrict__ W,
                                                 int M, int N, int K,
                                                 const float* __restrict__ resid) {
  __shared__ bf16 As[TM][LDP];
  __shared__ bf16 Ws[TN][LDP];
  int mbase = blockIdx.y * TM;
  int nbase = blockIdx.x * TN;
  int tid = threadIdx.x, lane = tid & 31, wave = tid >> 5;
  int wm = (wave >> 1) * 32, wn = (wave & 1) * 32;
  int ml = lane & 15, half = lane >> 4;
  v8f zero = {0.f, 0.f, 0.f, 0.f, 0.f, 0.f, 0.f, 0.f};
  v8f acc[2][2];
  for (int i = 0; i < 2; i++) for (int j = 0; j < 2; j++) acc[i][j] = zero;

  for (int kk = 0; kk < K; kk += TK) {
    // 64x32 bf16 tile per matrix = 256 16B-chunks; 128 threads x 2 iters
#pragma unroll
    for (int i = 0; i < 2; i++) {
      int chunk = i * 128 + tid;
      int r = chunk >> 2, c8 = (chunk & 3) * 8;
      int gm = mbase + r, gn = nbase + r;
      fill_lds_16B(&As[r][c8], A + (long)gm * K + kk + c8, gm < M);
      fill_lds_16B(&Ws[r][c8], W + (long)gn * K + kk + c8, gn < N);
    }
    wait_lds_fill();
    __syncthreads();
    v16bf af[2], bw[2];
#pragma unroll
    for (int t = 0; t < 2; t++) {
      af[t] = load_afrag(&As[wm + t * 16 + ml][0], half);
      bw[t] = load_bfrag(&Ws[wn + t * 16 + ml][0], half);
    }
#pragma unroll
    for (int i = 0; i < 2; i++)
#pragma unroll
      for (int j = 0; j < 2; j++)
        acc[i][j] = wmma_bf16(af[i], bw[j], acc[i][j]);
    __syncthreads();
  }
#pragma unroll
  for (int i = 0; i < 2; i++)
#pragma unroll
    for (int j = 0; j < 2; j++)
#pragma unroll
      for (int r = 0; r < 8; r++) {
        int gm = mbase + wm + i * 16 + half * 8 + r;
        int gn = nbase + wn + j * 16 + ml;
        if (gm < M && gn < N) {
          float v = acc[i][j][r];
          if (resid) v += resid[(long)gm * N + gn];
          C[(long)gm * N + gn] = v;
        }
      }
}

// ---- expert GEMM: per-expert runtime row count, optional token gather ------
__global__ __launch_bounds__(128) void k_gemm_expert(float* __restrict__ C,
                                                     const bf16* __restrict__ A,
                                                     const bf16* __restrict__ Wall,
                                                     int N, int K,
                                                     const int* __restrict__ elist,
                                                     const int* __restrict__ ecount,
                                                     const int* __restrict__ eoff,
                                                     int maxM, int gatherA) {
  __shared__ bf16 As[TM][LDP];
  __shared__ bf16 Ws[TN][LDP];
  int e    = blockIdx.z;
  int cnt  = ecount[e];
  int mbase = blockIdx.y * TM;
  if (mbase >= cnt) return;
  int nbase = blockIdx.x * TN;
  const bf16* W = Wall + (long)e * N * K;
  int base = eoff[e];
  int tid = threadIdx.x, lane = tid & 31, wave = tid >> 5;
  int wm = (wave >> 1) * 32, wn = (wave & 1) * 32;
  int ml = lane & 15, half = lane >> 4;
  v8f zero = {0.f, 0.f, 0.f, 0.f, 0.f, 0.f, 0.f, 0.f};
  v8f acc[2][2];
  for (int i = 0; i < 2; i++) for (int j = 0; j < 2; j++) acc[i][j] = zero;

  for (int kk = 0; kk < K; kk += TK) {
#pragma unroll
    for (int i = 0; i < 2; i++) {
      int chunk = i * 128 + tid;
      int r = chunk >> 2, c8 = (chunk & 3) * 8;
      int gm = mbase + r;
      bool ok = gm < cnt;
      int arow = gatherA ? (ok ? elist[e * maxM + gm] : 0) : (base + (ok ? gm : 0));
      fill_lds_16B(&As[r][c8], A + (long)arow * K + kk + c8, ok);
      int gn = nbase + r;
      fill_lds_16B(&Ws[r][c8], W + (long)gn * K + kk + c8, gn < N);
    }
    wait_lds_fill();
    __syncthreads();
    v16bf af[2], bw[2];
#pragma unroll
    for (int t = 0; t < 2; t++) {
      af[t] = load_afrag(&As[wm + t * 16 + ml][0], half);
      bw[t] = load_bfrag(&Ws[wn + t * 16 + ml][0], half);
    }
#pragma unroll
    for (int i = 0; i < 2; i++)
#pragma unroll
      for (int j = 0; j < 2; j++)
        acc[i][j] = wmma_bf16(af[i], bw[j], acc[i][j]);
    __syncthreads();
  }
#pragma unroll
  for (int i = 0; i < 2; i++)
#pragma unroll
    for (int j = 0; j < 2; j++)
#pragma unroll
      for (int r = 0; r < 8; r++) {
        int lm = mbase + wm + i * 16 + half * 8 + r;
        int gn = nbase + wn + j * 16 + ml;
        if (lm < cnt && gn < N)
          C[(long)(base + lm) * N + gn] = acc[i][j][r];
      }
}

// ---------------- flash attention: one wave per (b, h, 16-query tile) -------
// Q,K: (b, l, heads, 64) bf16 ; V: transposed (b, kv, 64, l) bf16.
__global__ __launch_bounds__(32) void k_flash_attn(float* __restrict__ O,
                                                   const bf16* __restrict__ Q,
                                                   const bf16* __restrict__ Kb,
                                                   const bf16* __restrict__ Vt,
                                                   int L) {
  int qt = blockIdx.x, h = blockIdx.y, b = blockIdx.z;
  int qbase = qt * 16;
  int hk = h / (cNH / cNKV);
  int lane = threadIdx.x;
  int ml = lane & 15, half = lane >> 4;
  const float scale = 0.125f;                         // 1/sqrt(64)

  // Q fragments (16x64 -> two 16x32 A-frags), loaded once, vectorized
  const bf16* qp = Q + (((long)b * L + qbase + ml) * cNH + h) * cHD;
  v16bf qf[2];
#pragma unroll
  for (int kc = 0; kc < 2; kc++) qf[kc] = load_afrag(qp + kc * 32, half);

  v8f zero = {0.f, 0.f, 0.f, 0.f, 0.f, 0.f, 0.f, 0.f};
  v8f o[4];
  for (int i = 0; i < 4; i++) o[i] = zero;
  float mprev[8], lsum[8];
  for (int r = 0; r < 8; r++) { mprev[r] = -1e30f; lsum[r] = 0.f; }

  __shared__ bf16 Pl[16][LDP];

  int kend = qbase + 16;
  for (int kb = 0; kb < kend; kb += 32) {
    v8f s[2];
    s[0] = zero; s[1] = zero;
#pragma unroll
    for (int t = 0; t < 2; t++) {
      const bf16* kp = Kb + (((long)b * L + kb + t * 16 + ml) * cNKV + hk) * cHD;
#pragma unroll
      for (int kc = 0; kc < 2; kc++) {
        v16bf kf = load_bfrag(kp + kc * 32, half);
        s[t] = wmma_bf16(qf[kc], kf, s[t]);
      }
    }
    // scale + causal mask in C layout: row = qbase+half*8+r, col = kb+t*16+ml
#pragma unroll
    for (int t = 0; t < 2; t++)
#pragma unroll
      for (int r = 0; r < 8; r++) {
        int q = qbase + half * 8 + r;
        int k = kb + t * 16 + ml;
        float v = s[t][r] * scale;
        s[t][r] = (k <= q) ? v : -1e30f;
      }
    // online softmax: row stats via 16-lane xor reductions
    float corr[8];
#pragma unroll
    for (int r = 0; r < 8; r++) {
      float rm = fmaxf(s[0][r], s[1][r]);
      for (int off = 1; off < 16; off <<= 1) rm = fmaxf(rm, __shfl_xor(rm, off, 32));
      float mnew = fmaxf(mprev[r], rm);
      corr[r] = __expf(mprev[r] - mnew);
      float p0 = __expf(s[0][r] - mnew);
      float p1 = __expf(s[1][r] - mnew);
      s[0][r] = p0; s[1][r] = p1;
      float rs = p0 + p1;
      for (int off = 1; off < 16; off <<= 1) rs += __shfl_xor(rs, off, 32);
      lsum[r] = lsum[r] * corr[r] + rs;
      mprev[r] = mnew;
    }
#pragma unroll
    for (int i = 0; i < 4; i++)
#pragma unroll
      for (int r = 0; r < 8; r++) o[i][r] *= corr[r];
    // re-layout P (C layout) -> A fragment via LDS
    __syncthreads();
#pragma unroll
    for (int t = 0; t < 2; t++)
#pragma unroll
      for (int r = 0; r < 8; r++)
        Pl[half * 8 + r][t * 16 + ml] = f2bf(s[t][r]);
    __syncthreads();
    v16bf pf = load_afrag(&Pl[ml][0], half);
    // P(16x32) x V(32x64) -> four 16x16 tiles; V transposed => contiguous keys
#pragma unroll
    for (int vt = 0; vt < 4; vt++) {
      const bf16* vp = Vt + (((long)b * cNKV + hk) * cHD + vt * 16 + ml) * L + kb;
      v16bf vf = load_bfrag(vp, half);
      o[vt] = wmma_bf16(pf, vf, o[vt]);
    }
  }
  // normalize and store (token, NH, HD) f32
#pragma unroll
  for (int r = 0; r < 8; r++) {
    float inv = 1.f / lsum[r];
    int q = qbase + half * 8 + r;
#pragma unroll
    for (int vt = 0; vt < 4; vt++)
      O[(((long)b * L + q) * cNH + h) * cHD + vt * 16 + ml] = o[vt][r] * inv;
  }
}

// ---------------- router + aux-loss accumulation ----------------
__global__ void k_router(const float* __restrict__ hn, const float* __restrict__ gw,
                         int* __restrict__ elist, float* __restrict__ ewt,
                         int* __restrict__ ecount, float* __restrict__ psum,
                         float* __restrict__ fcnt, float* __restrict__ zsum, int Ntok) {
  int n = blockIdx.x * blockDim.x + threadIdx.x;
  if (n >= Ntok) return;
  float lg[cNEXP];
  const float* xr = hn + (long)n * cD;
  for (int e = 0; e < cNEXP; e++) {
    const float* wr = gw + (long)e * cD;
    float s = 0.f;
    for (int k = 0; k < cD; k++) s += xr[k] * wr[k];
    lg[e] = s;
  }
  float mx = lg[0];
  for (int e = 1; e < cNEXP; e++) mx = fmaxf(mx, lg[e]);
  float ex[cNEXP], se = 0.f;
  for (int e = 0; e < cNEXP; e++) { ex[e] = __expf(lg[e] - mx); se += ex[e]; }
  float lse = mx + __logf(se);
  float sc[cNEXP];
  for (int e = 0; e < cNEXP; e++) sc[e] = ex[e] / se;
  int i0 = 0;
  for (int e = 1; e < cNEXP; e++) if (sc[e] > sc[i0]) i0 = e;
  int i1 = (i0 == 0) ? 1 : 0;
  for (int e = 0; e < cNEXP; e++) if (e != i0 && sc[e] > sc[i1]) i1 = e;
  float w0 = sc[i0], w1 = sc[i1], wsum = w0 + w1;
  w0 /= wsum; w1 /= wsum;
  for (int e = 0; e < cNEXP; e++) atomicAdd(&psum[e], sc[e]);
  atomicAdd(&fcnt[i0], 1.f);
  atomicAdd(&fcnt[i1], 1.f);
  atomicAdd(zsum, lse * lse);
  int p0 = atomicAdd(&ecount[i0], 1);
  elist[i0 * Ntok + p0] = n; ewt[i0 * Ntok + p0] = w0;
  int p1 = atomicAdd(&ecount[i1], 1);
  elist[i1 * Ntok + p1] = n; ewt[i1 * Ntok + p1] = w1;
}

__global__ void k_scan(const int* __restrict__ ecount, int* __restrict__ eoff) {
  if (threadIdx.x == 0 && blockIdx.x == 0) {
    int a = 0;
    for (int e = 0; e < cNEXP; e++) { eoff[e] = a; a += ecount[e]; }
  }
}

// ---- scatter expert output: moe[token] += w * eo[row] ----------------------
__global__ __launch_bounds__(256) void k_scatter(float* __restrict__ moe,
                                                 const float* __restrict__ eo,
                                                 const int* __restrict__ elist,
                                                 const float* __restrict__ ewt,
                                                 const int* __restrict__ ecount,
                                                 const int* __restrict__ eoff,
                                                 int maxM) {
  int e = blockIdx.y;
  int m = blockIdx.x;
  if (m >= ecount[e]) return;
  int token = elist[e * maxM + m];
  float w = ewt[e * maxM + m];
  const float* src = eo + (long)(eoff[e] + m) * cD;
  float* dst = moe + (long)token * cD;
  for (int c = threadIdx.x; c < cD; c += 256) atomicAdd(&dst[c], w * src[c]);
}

__global__ void k_loss(float* __restrict__ out, const float* __restrict__ psum,
                       const float* __restrict__ fcnt, const float* __restrict__ zsum,
                       float invN) {
  if (threadIdx.x == 0 && blockIdx.x == 0) {
    float a = 0.f;
    for (int e = 0; e < cNEXP; e++) a += (fcnt[e] * invN) * (psum[e] * invN);
    out[0] = 0.01f * (float)cNEXP * a + 0.001f * (zsum[0] * invN);
  }
}

// ---------------- host orchestration ----------------
static size_t aloc(size_t& off, size_t bytes) {
  size_t cur = off;
  off += (bytes + 255) & ~(size_t)255;
  return cur;
}

extern "C" void kernel_launch(void* const* d_in, const int* in_sizes, int n_in,
                              void* d_out, int out_size, void* d_ws, size_t ws_size,
                              hipStream_t stream) {
  (void)in_sizes; (void)n_in; (void)out_size; (void)ws_size;
  const float* x       = (const float*)d_in[0];
  const float* attn_nw = (const float*)d_in[1];
  const float* ffn_nw  = (const float*)d_in[2];
  const float* wq      = (const float*)d_in[3];
  const float* wk      = (const float*)d_in[4];
  const float* wv      = (const float*)d_in[5];
  const float* wo      = (const float*)d_in[6];
  const float* qnw     = (const float*)d_in[7];
  const float* knw     = (const float*)d_in[8];
  const float* gate_w  = (const float*)d_in[9];
  const float* eg      = (const float*)d_in[10];
  const float* eu      = (const float*)d_in[11];
  const float* ed      = (const float*)d_in[12];
  const float* shg     = (const float*)d_in[13];
  const float* shu     = (const float*)d_in[14];
  const float* shd     = (const float*)d_in[15];
  float* out = (float*)d_out;

  char* base = (char*)d_ws;
  size_t off = 0;
  const long KV = (long)cNKV * cHD;            // 256
  // bf16 weight copies
  bf16* wq_bf  = (bf16*)(base + aloc(off, (size_t)cD * cD * 2));
  bf16* wk_bf  = (bf16*)(base + aloc(off, (size_t)KV * cD * 2));
  bf16* wv_bf  = (bf16*)(base + aloc(off, (size_t)KV * cD * 2));
  bf16* wo_bf  = (bf16*)(base + aloc(off, (size_t)cD * cD * 2));
  bf16* shg_bf = (bf16*)(base + aloc(off, (size_t)cINTER * cD * 2));
  bf16* shu_bf = (bf16*)(base + aloc(off, (size_t)cINTER * cD * 2));
  bf16* shd_bf = (bf16*)(base + aloc(off, (size_t)cD * cINTER * 2));
  bf16* eg_bf  = (bf16*)(base + aloc(off, (size_t)cNEXP * cINTER * cD * 2));
  bf16* eu_bf  = (bf16*)(base + aloc(off, (size_t)cNEXP * cINTER * cD * 2));
  bf16* ed_bf  = (bf16*)(base + aloc(off, (size_t)cNEXP * cD * cINTER * 2));
  // activations
  bf16*  xn_bf = (bf16*)(base + aloc(off, (size_t)cN * cD * 2));
  float* qf    = (float*)(base + aloc(off, (size_t)cN * cD * 4));
  float* kf    = (float*)(base + aloc(off, (size_t)cN * KV * 4));
  float* vf    = (float*)(base + aloc(off, (size_t)cN * KV * 4));
  bf16*  qbf   = (bf16*)(base + aloc(off, (size_t)cN * cD * 2));
  bf16*  kbf   = (bf16*)(base + aloc(off, (size_t)cN * KV * 2));
  bf16*  vTbf  = (bf16*)(base + aloc(off, (size_t)cN * KV * 2));
  float* of    = (float*)(base + aloc(off, (size_t)cN * cD * 4));
  bf16*  obf   = (bf16*)(base + aloc(off, (size_t)cN * cD * 2));
  float* h     = (float*)(base + aloc(off, (size_t)cN * cD * 4));
  float* hnf   = (float*)(base + aloc(off, (size_t)cN * cD * 4));
  bf16*  hnbf  = (bf16*)(base + aloc(off, (size_t)cN * cD * 2));
  float* gsh   = (float*)(base + aloc(off, (size_t)cN * cINTER * 4));
  float* ush   = (float*)(base + aloc(off, (size_t)cN * cINTER * 4));
  bf16*  ash   = (bf16*)(base + aloc(off, (size_t)cN * cINTER * 2));
  float* gbuf  = (float*)(base + aloc(off, (size_t)cR * cINTER * 4));
  float* ubuf  = (float*)(base + aloc(off, (size_t)cR * cINTER * 4));
  bf16*  abuf  = (bf16*)(base + aloc(off, (size_t)cR * cINTER * 2));
  float* eobuf = (float*)(base + aloc(off, (size_t)cR * cD * 4));
  float* moe   = (float*)(base + aloc(off, (size_t)cN * cD * 4));
  int*   elist = (int*)(base + aloc(off, (size_t)cNEXP * cN * 4));
  float* ewt   = (float*)(base + aloc(off, (size_t)cNEXP * cN * 4));
  int*   ecnt  = (int*)(base + aloc(off, 64));
  int*   eoff  = (int*)(base + aloc(off, 64));
  float* psum  = (float*)(base + aloc(off, 64));
  float* fcnt  = (float*)(base + aloc(off, 64));
  float* zsum  = (float*)(base + aloc(off, 64));

  auto blks = [](long n, int t) { return (unsigned)((n + t - 1) / t); };

  // 1) weight conversions fp32 -> bf16
  k_cvt_bf16<<<blks((long)cD * cD, 256), 256, 0, stream>>>(wq_bf, wq, (long)cD * cD);
  k_cvt_bf16<<<blks(KV * cD, 256), 256, 0, stream>>>(wk_bf, wk, KV * cD);
  k_cvt_bf16<<<blks(KV * cD, 256), 256, 0, stream>>>(wv_bf, wv, KV * cD);
  k_cvt_bf16<<<blks((long)cD * cD, 256), 256, 0, stream>>>(wo_bf, wo, (long)cD * cD);
  k_cvt_bf16<<<blks((long)cINTER * cD, 256), 256, 0, stream>>>(shg_bf, shg, (long)cINTER * cD);
  k_cvt_bf16<<<blks((long)cINTER * cD, 256), 256, 0, stream>>>(shu_bf, shu, (long)cINTER * cD);
  k_cvt_bf16<<<blks((long)cD * cINTER, 256), 256, 0, stream>>>(shd_bf, shd, (long)cD * cINTER);
  k_cvt_bf16<<<blks((long)cNEXP * cINTER * cD, 256), 256, 0, stream>>>(eg_bf, eg, (long)cNEXP * cINTER * cD);
  k_cvt_bf16<<<blks((long)cNEXP * cINTER * cD, 256), 256, 0, stream>>>(eu_bf, eu, (long)cNEXP * cINTER * cD);
  k_cvt_bf16<<<blks((long)cNEXP * cD * cINTER, 256), 256, 0, stream>>>(ed_bf, ed, (long)cNEXP * cD * cINTER);

  // 2) attention-input RMSNorm
  k_rmsnorm<<<cN, 256, 0, stream>>>(xn_bf, (float*)nullptr, x, attn_nw, cD);

  // 3) q/k/v projections (WMMA)
  k_gemm_nt<<<dim3(cD / TN, cN / TM), 128, 0, stream>>>(qf, xn_bf, wq_bf, cN, cD, cD, nullptr);
  k_gemm_nt<<<dim3((int)KV / TN, cN / TM), 128, 0, stream>>>(kf, xn_bf, wk_bf, cN, (int)KV, cD, nullptr);
  k_gemm_nt<<<dim3((int)KV / TN, cN / TM), 128, 0, stream>>>(vf, xn_bf, wv_bf, cN, (int)KV, cD, nullptr);

  // 4) per-head RMSNorm + RoPE, then bf16 (V transposed for PV fragments)
  k_qknorm_rope<<<cN * cNH, 32, 0, stream>>>(qbf, qf, qnw, cNH, cL);
  k_qknorm_rope<<<cN * cNKV, 32, 0, stream>>>(kbf, kf, knw, cNKV, cL);
  k_cvt_v_t<<<blks(cN * KV, 256), 256, 0, stream>>>(vTbf, vf);

  // 5) causal flash attention (WMMA QK^T and PV)
  k_flash_attn<<<dim3(cL / 16, cNH, cB), 32, 0, stream>>>(of, qbf, kbf, vTbf, cL);

  // 6) output projection + residual
  k_cvt_bf16<<<blks((long)cN * cD, 256), 256, 0, stream>>>(obf, of, (long)cN * cD);
  k_gemm_nt<<<dim3(cD / TN, cN / TM), 128, 0, stream>>>(h, obf, wo_bf, cN, cD, cD, x);

  // 7) FFN-input RMSNorm (f32 for router, bf16 for GEMMs)
  k_rmsnorm<<<cN, 256, 0, stream>>>(hnbf, hnf, h, ffn_nw, cD);

  // 8) zero accumulators / routing state / moe output
  k_zero_i<<<1, 64, 0, stream>>>(ecnt, 8);
  k_zero_f<<<1, 64, 0, stream>>>(psum, 8);
  k_zero_f<<<1, 64, 0, stream>>>(fcnt, 8);
  k_zero_f<<<1, 64, 0, stream>>>(zsum, 1);
  k_zero_f<<<blks((long)cN * cD, 256), 256, 0, stream>>>(moe, (long)cN * cD);

  // 9) router + top-2 selection + aux-loss accumulation
  k_router<<<blks(cN, 256), 256, 0, stream>>>(hnf, gate_w, elist, ewt, ecnt, psum, fcnt, zsum, cN);
  k_scan<<<1, 32, 0, stream>>>(ecnt, eoff);

  // 10) routed experts: gate/up (gather), SiLU*up, down (WMMA)
  k_gemm_expert<<<dim3(cINTER / TN, cN / TM, cNEXP), 128, 0, stream>>>(
      gbuf, hnbf, eg_bf, cINTER, cD, elist, ecnt, eoff, cN, 1);
  k_gemm_expert<<<dim3(cINTER / TN, cN / TM, cNEXP), 128, 0, stream>>>(
      ubuf, hnbf, eu_bf, cINTER, cD, elist, ecnt, eoff, cN, 1);
  k_silu_mul<<<blks((long)cR * cINTER, 256), 256, 0, stream>>>(abuf, gbuf, ubuf, (long)cR * cINTER);
  k_gemm_expert<<<dim3(cD / TN, cN / TM, cNEXP), 128, 0, stream>>>(
      eobuf, abuf, ed_bf, cD, cINTER, elist, ecnt, eoff, cN, 0);
  k_scatter<<<dim3(cN, cNEXP), 256, 0, stream>>>(moe, eobuf, elist, ewt, ecnt, eoff, cN);

  // 11) shared expert
  k_gemm_nt<<<dim3(cINTER / TN, cN / TM), 128, 0, stream>>>(gsh, hnbf, shg_bf, cN, cINTER, cD, nullptr);
  k_gemm_nt<<<dim3(cINTER / TN, cN / TM), 128, 0, stream>>>(ush, hnbf, shu_bf, cN, cINTER, cD, nullptr);
  k_silu_mul<<<blks((long)cN * cINTER, 256), 256, 0, stream>>>(ash, gsh, ush, (long)cN * cINTER);
  k_gemm_nt<<<dim3(cD / TN, cN / TM), 128, 0, stream>>>(moe, ash, shd_bf, cN, cD, cINTER, moe);

  // 12) final residual + loss scalar
  k_add_out<<<blks((long)cN * cD, 256), 256, 0, stream>>>(out, h, moe, (long)cN * cD);
  k_loss<<<1, 32, 0, stream>>>(out + (long)cN * cD, psum, fcnt, zsum, 1.f / (float)cN);
}
